// BertEmbedding_34909494182296
// MI455X (gfx1250) — matrix operand order repaired and checked by reading, not compiled, not run on hardware
//
#include <hip/hip_runtime.h>
#include <hip/hip_bf16.h>

// Problem constants (from reference): L=13 layers (use 1..12), B=32, T=512, H=768
#define BB    32
#define TT_   512
#define HH    768
#define NL    12      // layers used
#define TPB   256     // threads per block (8 wave32)
#define HPT   3       // hidden dims per thread: 768/256
#define TOKT  16      // tokens per block tile
#define TILES (TT_/TOKT)   // 32 tiles per batch row

typedef __attribute__((ext_vector_type(16))) _Float16 v16h;
typedef __attribute__((ext_vector_type(8)))  float    v8f;

// ---------------------------------------------------------------------------
// Kernel 1: streaming pass over hidden_states (603 MB -> HBM-bound, ~26us at
// 23.3 TB/s). Per token: mean over 12 layers, 12 score dot-products,
// softmax(12), weighted recombination, running masked max over tokens.
// grid = (B, TILES), block = 256 (8 wave32)
// ---------------------------------------------------------------------------
__global__ __launch_bounds__(TPB) void bissect_pool_kernel(
    const float* __restrict__ hs,     // (13, B, T, H)
    const float* __restrict__ mask,   // (B, T)
    float* __restrict__ part)         // (B, TILES, H)
{
    const int b    = blockIdx.x;
    const int tile = blockIdx.y;
    const int tid  = threadIdx.x;
    const int wave = tid >> 5;
    const int lane = tid & 31;

    __shared__ float sred[TPB / 32][NL];

    float runmax[HPT];
#pragma unroll
    for (int j = 0; j < HPT; ++j) runmax[j] = -3.402823466e38f;

    const size_t layer_stride = (size_t)BB * TT_ * HH;

    for (int tt = 0; tt < TOKT; ++tt) {
        const int t = tile * TOKT + tt;
        const size_t tok_off = ((size_t)b * TT_ + t) * (size_t)HH;

        // Load 12 layers x 3 dims into registers; NT hint (stream-once data,
        // 603 MB > 192 MB L2 so avoid polluting it).
        float hsv[NL][HPT];
        float vsum[HPT] = {0.f, 0.f, 0.f};
#pragma unroll
        for (int l = 0; l < NL; ++l) {
            const float* p = hs + (size_t)(l + 1) * layer_stride + tok_off;
#pragma unroll
            for (int j = 0; j < HPT; ++j) {
                float x = __builtin_nontemporal_load(p + tid + j * TPB);
                hsv[l][j] = x;
                vsum[j] += x;
            }
        }
        float v[HPT];
#pragma unroll
        for (int j = 0; j < HPT; ++j) v[j] = vsum[j] * (1.0f / NL);

        // Per-thread partial score dot-products, wave32 xor-shuffle reduction.
        float ps[NL];
#pragma unroll
        for (int n = 0; n < NL; ++n) {
            float s = 0.f;
#pragma unroll
            for (int j = 0; j < HPT; ++j) s += v[j] * hsv[n][j];
#pragma unroll
            for (int off = 16; off > 0; off >>= 1)
                s += __shfl_xor(s, off, 32);
            ps[n] = s;
        }
        if (lane == 0) {
#pragma unroll
            for (int n = 0; n < NL; ++n) sred[wave][n] = ps[n];
        }
        __syncthreads();
        float score[NL];
#pragma unroll
        for (int n = 0; n < NL; ++n) {
            float s = 0.f;
#pragma unroll
            for (int w = 0; w < TPB / 32; ++w) s += sred[w][n];
            score[n] = s;
        }
        __syncthreads();   // sred reused next token

        // softmax over 12 (redundant per thread; trivial)
        float mx = score[0];
#pragma unroll
        for (int n = 1; n < NL; ++n) mx = fmaxf(mx, score[n]);
        float e[NL];
        float se = 0.f;
#pragma unroll
        for (int n = 0; n < NL; ++n) { e[n] = __expf(score[n] - mx); se += e[n]; }
        const float inv = 1.0f / se;

        const float mv = mask[b * TT_ + t];
#pragma unroll
        for (int j = 0; j < HPT; ++j) {
            float f = 0.f;
#pragma unroll
            for (int n = 0; n < NL; ++n) f += hsv[n][j] * e[n];
            runmax[j] = fmaxf(runmax[j], f * inv * mv);
        }
    }

    float* dst = part + ((size_t)b * TILES + tile) * (size_t)HH;
#pragma unroll
    for (int j = 0; j < HPT; ++j) dst[tid + j * TPB] = runmax[j];
}

// ---------------------------------------------------------------------------
// Kernel 2: reduce partial maxes over tiles -> pooled (B, H)
// ---------------------------------------------------------------------------
__global__ __launch_bounds__(TPB) void tile_max_kernel(
    const float* __restrict__ part, float* __restrict__ pooled)
{
    const int idx = blockIdx.x * TPB + threadIdx.x;   // over B*H
    if (idx >= BB * HH) return;
    const int b = idx / HH;
    const int h = idx - b * HH;
    const float* p = part + (size_t)b * TILES * HH + h;
    float m = -3.402823466e38f;
#pragma unroll
    for (int i = 0; i < TILES; ++i) m = fmaxf(m, p[(size_t)i * HH]);
    pooled[idx] = m;
}

// ---------------------------------------------------------------------------
// Kernel 3: logits = pooled(32x768) @ W^T(768x2) + bias, via WMMA.
// Single wave32. Two 16-row M tiles, K swept in chunks of 32 with
// v_wmma_f32_16x16x32_f16. Layouts per CDNA5 ISA 7.12.2 (hi = lane>>4):
//   A 16x32 f16 : lane supplies row (lane&15);
//                 halves 0..7  <- K = 8*hi + 0..7       (contiguous)
//                 halves 8..15 <- K = 16 + 8*hi + 0..7  (contiguous)
//   B 32x16 f16 : lane supplies col (lane&15); half p <- same K as A half p
//   D 16x16 f32 : lane col (lane&15); VGPR r -> row = 8*hi + r
// All fragment loads are branchless b128s; lanes with col>=2 read a valid
// clamped W row and select 0 afterwards (no EXEC manipulation in the loop).
// ---------------------------------------------------------------------------
__global__ __launch_bounds__(32) void logits_wmma_kernel(
    const float* __restrict__ pooled,  // (B, H)
    const float* __restrict__ Wm,      // (2, H)
    const float* __restrict__ bias,    // (2,)
    float* __restrict__ out)           // (B, 2)
{
    const int lane = threadIdx.x;
    const int nl   = lane & 15;        // A row within tile / B col / D col
    const int hi   = lane >> 4;
    const float bsel = (nl < 2) ? 1.0f : 0.0f;   // branchless B masking
    const int   wrow = nl & 1;                   // clamped valid W row

#pragma unroll
    for (int mtile = 0; mtile < 2; ++mtile) {
        v8f c = {};
        const int arow = mtile * 16 + nl;
        for (int k0 = 0; k0 < HH; k0 += 32) {
            const float4* ap0 = (const float4*)(pooled + arow * HH + k0 + hi * 8);
            const float4* ap1 = (const float4*)(pooled + arow * HH + k0 + 16 + hi * 8);
            const float4* bp  = (const float4*)(Wm + wrow * HH + k0 + hi * 8);
            const float4* bp1 = (const float4*)(Wm + wrow * HH + k0 + 16 + hi * 8);
            float4 a01 = ap0[0], a23 = ap0[1];     // A halves 0..7
            float4 a45 = ap1[0], a67 = ap1[1];     // A halves 8..15
            float4 b0 = bp[0], b1 = bp[1], b2 = bp1[0], b3 = bp1[1];

            v16h a, bm;
            a[0]  = (_Float16)a01.x; a[1]  = (_Float16)a01.y;
            a[2]  = (_Float16)a01.z; a[3]  = (_Float16)a01.w;
            a[4]  = (_Float16)a23.x; a[5]  = (_Float16)a23.y;
            a[6]  = (_Float16)a23.z; a[7]  = (_Float16)a23.w;
            a[8]  = (_Float16)a45.x; a[9]  = (_Float16)a45.y;
            a[10] = (_Float16)a45.z; a[11] = (_Float16)a45.w;
            a[12] = (_Float16)a67.x; a[13] = (_Float16)a67.y;
            a[14] = (_Float16)a67.z; a[15] = (_Float16)a67.w;

            bm[0]  = (_Float16)(b0.x * bsel); bm[1]  = (_Float16)(b0.y * bsel);
            bm[2]  = (_Float16)(b0.z * bsel); bm[3]  = (_Float16)(b0.w * bsel);
            bm[4]  = (_Float16)(b1.x * bsel); bm[5]  = (_Float16)(b1.y * bsel);
            bm[6]  = (_Float16)(b1.z * bsel); bm[7]  = (_Float16)(b1.w * bsel);
            bm[8]  = (_Float16)(b2.x * bsel); bm[9]  = (_Float16)(b2.y * bsel);
            bm[10] = (_Float16)(b2.z * bsel); bm[11] = (_Float16)(b2.w * bsel);
            bm[12] = (_Float16)(b3.x * bsel); bm[13] = (_Float16)(b3.y * bsel);
            bm[14] = (_Float16)(b3.z * bsel); bm[15] = (_Float16)(b3.w * bsel);

            // (neg_a, A, neg_b, B, c_mod, C, reuse_a, reuse_b)
            c = __builtin_amdgcn_wmma_f32_16x16x32_f16(
                    false, a, false, bm, (short)0, c, false, false);
        }
#pragma unroll
        for (int r = 0; r < 8; ++r) {
            const int M = mtile * 16 + hi * 8 + r;   // output row
            if (nl < 2) out[M * 2 + nl] = c[r] + bias[nl];
        }
    }
}

// ---------------------------------------------------------------------------
extern "C" void kernel_launch(void* const* d_in, const int* in_sizes, int n_in,
                              void* d_out, int out_size, void* d_ws, size_t ws_size,
                              hipStream_t stream) {
    const float* hs   = (const float*)d_in[0];  // (13, 32, 512, 768) fp32
    const float* mask = (const float*)d_in[1];  // (32, 512) fp32
    const float* Wm   = (const float*)d_in[2];  // (2, 768) fp32
    const float* bias = (const float*)d_in[3];  // (2,) fp32
    float* out = (float*)d_out;                 // (32, 2) fp32

    float* part   = (float*)d_ws;                         // B*TILES*H floats (~3 MB)
    float* pooled = part + (size_t)BB * TILES * HH;       // B*H floats

    dim3 g1(BB, TILES);
    bissect_pool_kernel<<<g1, TPB, 0, stream>>>(hs, mask, part);

    const int n2 = BB * HH;
    tile_max_kernel<<<(n2 + TPB - 1) / TPB, TPB, 0, stream>>>(part, pooled);

    logits_wmma_kernel<<<1, 32, 0, stream>>>(pooled, Wm, bias, out);
}
